// NPairLoss_68959994904848
// MI455X (gfx1250) — matrix operand order repaired and checked by reading, not compiled, not run on hardware
//
#include <hip/hip_runtime.h>
#include <hip/hip_bf16.h>

typedef __attribute__((ext_vector_type(16))) _Float16 v16h;  // 8 VGPRs
typedef __attribute__((ext_vector_type(8)))  _Float16 v8h;   // 4 VGPRs
typedef __attribute__((ext_vector_type(8)))  float    v8f;   // 8 VGPRs

#define DIM   512
#define KSTEP 32
#define NFRAG (DIM / KSTEP)          // 16 K-fragments of 32
#define BN    16                     // columns per LDS-staged B tile
#define LDS_ROW_BYTES 1040           // 1024 data + 16 pad (kills ds bank conflicts)
#define TILE_LDS_BYTES (BN * LDS_ROW_BYTES)

// ---------------------------------------------------------------------------
// Kernel 1: L2-normalize each row of embs (f32) -> eh (f16), zero pos/tot.
// ---------------------------------------------------------------------------
__global__ __launch_bounds__(256) void norm_kernel(const float* __restrict__ embs,
                                                   _Float16* __restrict__ eh,
                                                   float* __restrict__ pos,
                                                   float* __restrict__ tot) {
    const int r = blockIdx.x;
    const int t = threadIdx.x;
    const size_t base = (size_t)r * DIM;
    float v0 = embs[base + t];
    float v1 = embs[base + 256 + t];
    float ss = v0 * v0 + v1 * v1;
    #pragma unroll
    for (int m = 1; m < 32; m <<= 1) ss += __shfl_xor(ss, m, 32);
    __shared__ float red[8];
    if ((t & 31) == 0) red[t >> 5] = ss;
    __syncthreads();
    float total = 0.0f;
    #pragma unroll
    for (int w = 0; w < 8; ++w) total += red[w];
    const float inv = rsqrtf(total);
    eh[base + t]       = (_Float16)(v0 * inv);
    eh[base + 256 + t] = (_Float16)(v1 * inv);
    if (t == 0) { pos[r] = 0.0f; tot[r] = 0.0f; }
}

// ---------------------------------------------------------------------------
// Async-DMA a 16-column B panel (16 rows x 1024B) into padded LDS rows.
// 128 threads x 8 chunks x 16B = 16 KB. Tracked by ASYNCcnt.
// ---------------------------------------------------------------------------
__device__ __forceinline__ void stage_tile_async(const _Float16* __restrict__ eh,
                                                 int j0,
                                                 unsigned char* smemBase,
                                                 int tid) {
    const uint64_t gbase = (uint64_t)(uintptr_t)(eh + (size_t)j0 * DIM);
    #pragma unroll
    for (int c = 0; c < 8; ++c) {
        const int ch  = tid + c * 128;               // 0..1023 16B chunks
        const int row = ch >> 6;                     // 0..15 (column index)
        const int off = (ch & 63) << 4;              // 0..1008 within row
        const uint32_t lds = (uint32_t)(uintptr_t)(smemBase + row * LDS_ROW_BYTES + off);
        const uint64_t g   = gbase + (uint64_t)row * (DIM * 2) + (uint64_t)off;
        asm volatile("global_load_async_to_lds_b128 %0, %1, off"
                     :: "v"(lds), "v"(g) : "memory");
    }
}

// ---------------------------------------------------------------------------
// Kernel 2: WMMA Gram-matrix strips. Block = 4 waves = 64 rows. All waves
// share each 16-column B panel through double-buffered LDS (async copies
// overlap the next panel's DMA with the current panel's 16-deep WMMA chain).
// A panel (16x512 f16 per wave) stays register-resident.
// ---------------------------------------------------------------------------
__global__ __launch_bounds__(128) void sim_kernel(const _Float16* __restrict__ eh,
                                                  const int* __restrict__ labels,
                                                  float* __restrict__ pos,
                                                  float* __restrict__ tot,
                                                  int jChunk) {
    __shared__ __align__(16) unsigned char smem[2 * TILE_LDS_BYTES];

    const int tid     = threadIdx.x;      // 0..127
    const int wid     = tid >> 5;         // wave 0..3
    const int lane    = tid & 31;
    const int half    = lane >> 4;        // lane half 0|1
    const int l16     = lane & 15;
    const int rowBase = blockIdx.x * 64 + wid * 16;
    const int jBegin  = blockIdx.y * jChunk;
    const int nTiles  = jChunk / BN;

    // ---- Preload A fragments (contiguous K order, matching B):
    // lane m (<16):  v0..3 = K+0..7  (+0B),  v4..7 = K+8..15 (+16B)
    // lane m+16:     v0..3 = K+16..23 (+32B), v4..7 = K+24..31 (+48B)
    v16h aF[NFRAG];
    {
        const _Float16* arow = eh + (size_t)(rowBase + l16) * DIM + half * 16;
        #pragma unroll
        for (int kk = 0; kk < NFRAG; ++kk) {
            v8h lo = *(const v8h*)(arow + kk * KSTEP);
            v8h hi = *(const v8h*)(arow + kk * KSTEP + 8);
            v16h f;
            #pragma unroll
            for (int i = 0; i < 8; ++i) { f[i] = lo[i]; f[i + 8] = hi[i]; }
            aF[kk] = f;
        }
    }

    int rlab[8];
    #pragma unroll
    for (int r = 0; r < 8; ++r) rlab[r] = labels[rowBase + half * 8 + r];

    float pacc[8], tacc[8];
    #pragma unroll
    for (int r = 0; r < 8; ++r) { pacc[r] = 0.0f; tacc[r] = 0.0f; }

    // Prologue: DMA tile 0 into buffer 0.
    stage_tile_async(eh, jBegin, smem, tid);

    #pragma unroll 1
    for (int t = 0; t < nTiles; ++t) {
        const int j0 = jBegin + t * BN;

        // My async copies for tile t have landed; everyone done with the
        // buffer we are about to overwrite (they read it in iteration t-1).
        asm volatile("s_wait_asynccnt 0x0" ::: "memory");
        __syncthreads();

        // Kick off DMA for tile t+1 into the other buffer; overlaps compute.
        if (t + 1 < nTiles)
            stage_tile_async(eh, j0 + BN, smem + ((t + 1) & 1) * TILE_LDS_BYTES, tid);

        // ---- 16-deep WMMA chain, B fragments from LDS.
        // B 32x16 layout: lane n(<16): col n, K k0..k0+15 ; lane n+16: K k0+16..31.
        const unsigned char* bbase =
            smem + (t & 1) * TILE_LDS_BYTES + l16 * LDS_ROW_BYTES + half * 32;
        v8f acc = {};
        #pragma unroll
        for (int kk = 0; kk < NFRAG; ++kk) {
            v16h bf = *(const v16h*)(bbase + kk * 64);
            acc = __builtin_amdgcn_wmma_f32_16x16x32_f16(
                false, aF[kk], false, bf, (short)0, acc, false, false);
        }

        // ---- Masked exp-accumulate. C/D layout: M = r + 8*half, N = l16.
        const int col  = j0 + l16;
        const int clab = labels[col];
        #pragma unroll
        for (int r = 0; r < 8; ++r) {
            const int row = rowBase + half * 8 + r;
            const float e = __expf(acc[r]);          // tau == 1
            const bool offd = (row != col);
            tacc[r] += offd ? e : 0.0f;
            pacc[r] += (offd && (clab == rlab[r])) ? e : 0.0f;
        }
    }

    // Reduce each row's 16 N-lanes (per half), one atomic per row per array.
    #pragma unroll
    for (int r = 0; r < 8; ++r) {
        float p = pacc[r], t = tacc[r];
        #pragma unroll
        for (int m = 1; m < 16; m <<= 1) {
            p += __shfl_xor(p, m, 32);
            t += __shfl_xor(t, m, 32);
        }
        if (l16 == 0) {
            const int row = rowBase + half * 8 + r;
            atomicAdd(&pos[row], p);
            atomicAdd(&tot[row], t);
        }
    }
}

// ---------------------------------------------------------------------------
// Kernel 3: label histogram (validity) + final log-ratio mean. One block.
// ---------------------------------------------------------------------------
__global__ __launch_bounds__(256) void loss_kernel(const int* __restrict__ labels,
                                                   const float* __restrict__ pos,
                                                   const float* __restrict__ tot,
                                                   float* __restrict__ out,
                                                   int N, int C) {
    __shared__ int   cnt[1024];
    __shared__ float sred[256];
    __shared__ int   nred[256];
    const int t = threadIdx.x;
    for (int c = t; c < C; c += 256) cnt[c] = 0;
    __syncthreads();
    for (int i = t; i < N; i += 256) atomicAdd(&cnt[labels[i] & (C - 1)], 1);
    __syncthreads();
    float sum = 0.0f; int nv = 0;
    for (int i = t; i < N; i += 256) {
        if (cnt[labels[i] & (C - 1)] > 1) {
            sum += __logf(tot[i]) - __logf(pos[i]);
            ++nv;
        }
    }
    sred[t] = sum; nred[t] = nv;
    __syncthreads();
    for (int s = 128; s > 0; s >>= 1) {
        if (t < s) { sred[t] += sred[t + s]; nred[t] += nred[t + s]; }
        __syncthreads();
    }
    if (t == 0) out[0] = (nred[0] > 0) ? (sred[0] / (float)nred[0]) : 0.0f;
}

// ---------------------------------------------------------------------------
extern "C" void kernel_launch(void* const* d_in, const int* in_sizes, int n_in,
                              void* d_out, int out_size, void* d_ws, size_t ws_size,
                              hipStream_t stream) {
    const float* embs   = (const float*)d_in[0];
    const int*   labels = (const int*)d_in[1];
    float*       out    = (float*)d_out;

    const int N = in_sizes[1];        // 8192
    const int C = 1024;

    _Float16* eh  = (_Float16*)d_ws;
    float*    pos = (float*)((char*)d_ws + (size_t)N * DIM * sizeof(_Float16));
    float*    tot = pos + N;

    norm_kernel<<<N, 256, 0, stream>>>(embs, eh, pos, tot);

    const int JSPLIT = 16;
    sim_kernel<<<dim3(N / 64, JSPLIT), 128, 0, stream>>>(eh, labels, pos, tot, N / JSPLIT);

    loss_kernel<<<1, 256, 0, stream>>>(labels, pos, tot, out, N, C);
}
